// InferCode_22651657519716
// MI455X (gfx1250) — matrix operand order repaired and checked by reading, not compiled, not run on hardware
//
#include <hip/hip_runtime.h>
#include <hip/hip_bf16.h>

typedef __attribute__((ext_vector_type(16))) _Float16 v16h;
typedef __attribute__((ext_vector_type(8)))  float    v8f;

#define WMMA_F16(a, b, c) \
  __builtin_amdgcn_wmma_f32_16x16x32_f16(false, (a), false, (b), (short)0, (c), false, false)

// ---------- helpers ----------

// fast f32 global atomic add (no return) -> global_atomic_add_f32
static __device__ inline void atom_add_f32(float* p, float v) {
  asm volatile("global_atomic_add_f32 %0, %1, off"
               :: "v"((unsigned long long)(size_t)p), "v"(v) : "memory");
}

// order-preserving float<->uint encoding for atomic max on floats
static __device__ inline unsigned fenc(float x) {
  unsigned u = __float_as_uint(x);
  return (u & 0x80000000u) ? ~u : (u | 0x80000000u);
}
static __device__ inline float fdec(unsigned k) {
  unsigned u = (k & 0x80000000u) ? (k & 0x7fffffffu) : ~k;
  return __uint_as_float(u);
}

// A fragment (16x32 f16): 16 consecutive k values per lane
static __device__ inline v16h pack_a_f32(const float* src) {
  v16h a;
#pragma unroll
  for (int j = 0; j < 8; ++j) { a[j] = (_Float16)src[j]; a[j + 8] = (_Float16)src[j + 8]; }
  return a;
}
static __device__ inline v16h pack_a_f16(const _Float16* src) {
  v16h a;
#pragma unroll
  for (int j = 0; j < 8; ++j) { a[j] = src[j]; a[j + 8] = src[j + 8]; }
  return a;
}
// B fragment (32x16 f16): 16 consecutive k values per lane
static __device__ inline v16h pack_b_f32(const float* src) {
  v16h b;
#pragma unroll
  for (int j = 0; j < 16; ++j) b[j] = (_Float16)src[j];
  return b;
}

// ---------- kernel 0: zero fill ----------
__global__ void kzero(float* p, int n) {
  int i = blockIdx.x * blockDim.x + threadIdx.x;
  int stride = gridDim.x * blockDim.x;
  for (; i < n; i += stride) p[i] = 0.0f;
}

// ---------- kernel 1: encoder GEMM + eta-weighted conv GEMM + scatter-add ----------
// block = 256 threads = 8 waves; each wave owns 16 nodes.
// dynamic LDS: Wenc f16 [128][256] (64KB) | Wstk f16 [384][128] (96KB) | Hbuf f16 [8][16][128] (32KB)
__global__ void k1_conv(const int* __restrict__ type_batch, const int* __restrict__ token_batch,
                        const int* __restrict__ node_indices,
                        const float* __restrict__ eta_t, const float* __restrict__ eta_l,
                        const float* __restrict__ eta_r,
                        const float* __restrict__ emb_type, const float* __restrict__ emb_token,
                        const float* __restrict__ W_enc, const float* __restrict__ b_enc,
                        const float* __restrict__ w_t, const float* __restrict__ w_l,
                        const float* __restrict__ w_r,
                        float* __restrict__ pre, int Nv) {
  extern __shared__ char smem[];
  _Float16* Wenc = (_Float16*)smem;                    // 128*256 halves
  _Float16* Wstk = (_Float16*)(smem + 65536);          // 384*128 halves
  _Float16* Hbuf = (_Float16*)(smem + 65536 + 98304);  // 8*16*128 halves

  __shared__ int   s_nidx[8][16];
  __shared__ float s_etaT[8][16], s_etaL[8][16], s_etaR[8][16];

  const int tid  = threadIdx.x;
  const int wave = tid >> 5;
  const int lane = tid & 31;
  const int m    = lane & 15;
  const int hi   = lane >> 4;
  const int kb2  = hi * 16;
  const int nodeBase = (blockIdx.x * 8 + wave) * 16;

  // stage W_enc (f32->f16) into LDS
  for (int i = tid; i < 128 * 256; i += 256) Wenc[i] = (_Float16)W_enc[i];
  // stage stacked [w_t; w_l; w_r] as rows [384][128]
  for (int i = tid; i < 384 * 128; i += 256) {
    int row = i >> 7, k = i & 127;
    float v = (row < 128) ? w_t[row * 128 + k]
            : (row < 256) ? w_l[(row - 128) * 128 + k]
                          : w_r[(row - 256) * 128 + k];
    Wstk[i] = (_Float16)v;
  }
  // stage per-node metadata
  if (lane < 16) {
    int node = nodeBase + lane;
    bool ok = node < Nv;
    int nc = ok ? node : 0;
    s_nidx[wave][lane] = ok ? node_indices[nc] : -1;
    s_etaT[wave][lane] = ok ? eta_t[nc] : 0.0f;
    s_etaL[wave][lane] = ok ? eta_l[nc] : 0.0f;
    s_etaR[wave][lane] = ok ? eta_r[nc] : 0.0f;
  }
  __syncthreads();

  // gather node features (concat[type_emb, token_emb]) as A fragments, K=256 (8 chunks)
  int node = nodeBase + m;
  int nc = node < Nv ? node : (Nv - 1);
  const float* trow = emb_type  + (long)type_batch[nc]  * 128;
  const float* krow = emb_token + (long)token_batch[nc] * 128;

  v16h af[8];
#pragma unroll
  for (int kc = 0; kc < 8; ++kc) {
    int kb = kc * 32 + hi * 16;
    const float* src = (kb < 128) ? (trow + kb) : (krow + (kb - 128));
    af[kc] = pack_a_f32(src);
  }

  // GEMM1: hidden[16x128] = cat[16x256] @ W_enc^T ; spill hidden (+b_enc) to LDS as f16
  _Float16* hb = Hbuf + wave * 2048;
  for (int nt = 0; nt < 8; ++nt) {
    v8f acc = {};
#pragma unroll
    for (int kc = 0; kc < 8; ++kc) {
      v16h b = *(const v16h*)(Wenc + (nt * 16 + m) * 256 + kc * 32 + kb2);
      acc = WMMA_F16(af[kc], b, acc);
    }
    int d = nt * 16 + m;              // lane holds column d of C tile
    float be = b_enc[d];
#pragma unroll
    for (int r = 0; r < 8; ++r) hb[(r + 8 * hi) * 128 + d] = (_Float16)(acc[r] + be);
  }
  asm volatile("s_wait_dscnt 0" ::: "memory");  // wave-local LDS transpose fence

  // reload hidden as A fragments, K=128 (4 chunks)
  v16h hf[4];
#pragma unroll
  for (int kc = 0; kc < 4; ++kc) hf[kc] = pack_a_f16(hb + m * 128 + kc * 32 + hi * 16);

  // per-lane row metadata (8 rows of the C tile)
  float etT[8], etL[8], etR[8]; int nix[8];
#pragma unroll
  for (int r = 0; r < 8; ++r) {
    int m2 = r + 8 * hi;
    nix[r] = s_nidx[wave][m2];
    etT[r] = s_etaT[wave][m2]; etL[r] = s_etaL[wave][m2]; etR[r] = s_etaR[wave][m2];
  }

  // GEMM2: conv tiles for w_t/w_l/w_r, combine with etas, scatter-add to pre
  for (int nt = 0; nt < 8; ++nt) {
    v8f aT = {}, aL = {}, aR = {};
#pragma unroll
    for (int kc = 0; kc < 4; ++kc) {
      v16h bT = *(const v16h*)(Wstk + ((nt      ) * 16 + m) * 128 + kc * 32 + kb2);
      v16h bL = *(const v16h*)(Wstk + ((nt +  8) * 16 + m) * 128 + kc * 32 + kb2);
      v16h bR = *(const v16h*)(Wstk + ((nt + 16) * 16 + m) * 128 + kc * 32 + kb2);
      aT = WMMA_F16(hf[kc], bT, aT);
      aL = WMMA_F16(hf[kc], bL, aL);
      aR = WMMA_F16(hf[kc], bR, aR);
    }
    int d = nt * 16 + m;
#pragma unroll
    for (int r = 0; r < 8; ++r) {
      if (nix[r] >= 0) {
        float v = etT[r] * aT[r] + etL[r] * aL[r] + etR[r] * aR[r];
        atom_add_f32(pre + (long)nix[r] * 128 + d, v);
      }
    }
  }
}

// ---------- kernel 2: node_emb = tanh(pre + bias); score = emb . alpha; tree max ----------
__global__ void k2_act(float* __restrict__ pre, float* __restrict__ scores,
                       unsigned* __restrict__ treeMax, const int* __restrict__ tree_indices,
                       const float* __restrict__ alpha, const float* __restrict__ bias_conv,
                       int NNv) {
  int wave = threadIdx.x >> 5, lane = threadIdx.x & 31;
  int node = blockIdx.x * 8 + wave;
  if (node >= NNv) return;
  float b = bias_conv[0];
  float4 x = *(float4*)(pre + (long)node * 128 + lane * 4);
  x.x = tanhf(x.x + b); x.y = tanhf(x.y + b); x.z = tanhf(x.z + b); x.w = tanhf(x.w + b);
  *(float4*)(pre + (long)node * 128 + lane * 4) = x;  // pre becomes node_emb in place
  float4 al = *(const float4*)(alpha + lane * 4);
  float s = x.x * al.x + x.y * al.y + x.z * al.z + x.w * al.w;
#pragma unroll
  for (int off = 16; off > 0; off >>= 1) s += __shfl_xor(s, off, 32);
  if (lane == 0) {
    scores[node] = s;
    atomicMax(treeMax + tree_indices[node], fenc(s));
  }
}

// ---------- kernel 3: softmax numerator/denominator accumulation ----------
__global__ void k3_pool(const float* __restrict__ node_emb, const float* __restrict__ scores,
                        const unsigned* __restrict__ treeMax, float* __restrict__ denom,
                        float* __restrict__ codeAcc, const int* __restrict__ tree_indices,
                        int NNv) {
  int wave = threadIdx.x >> 5, lane = threadIdx.x & 31;
  int node = blockIdx.x * 8 + wave;
  if (node >= NNv) return;
  int tr = tree_indices[node];
  float e = __expf(scores[node] - fdec(treeMax[tr]));
  float4 x = *(const float4*)(node_emb + (long)node * 128 + lane * 4);
  float* dst = codeAcc + tr * 128 + lane * 4;
  atom_add_f32(dst + 0, x.x * e);
  atom_add_f32(dst + 1, x.y * e);
  atom_add_f32(dst + 2, x.z * e);
  atom_add_f32(dst + 3, x.w * e);
  if (lane == 0) atom_add_f32(denom + tr, e);
}

// ---------- kernel 4: normalize code vectors, convert to f16 ----------
__global__ void k4_norm(const float* __restrict__ codeAcc, const float* __restrict__ denom,
                        _Float16* __restrict__ codeh, int n) {
  int i = blockIdx.x * blockDim.x + threadIdx.x;
  if (i < n) codeh[i] = (_Float16)(codeAcc[i] / denom[i >> 7]);
}

// ---------- kernel 5: logits = code @ W_out^T + b_out via WMMA ----------
__global__ void k5_logits(const _Float16* __restrict__ codeh, const float* __restrict__ W_out,
                          const float* __restrict__ b_out, float* __restrict__ out,
                          int S, int Tv) {
  int wave = threadIdx.x >> 5, lane = threadIdx.x & 31;
  int nt = blockIdx.x * 8 + wave;
  if (nt * 16 >= S) return;
  int n = lane & 15, hi = lane >> 4;
  int col = nt * 16 + n;
  const float* wrow = W_out + (long)col * 128;

  v16h bf[4];
#pragma unroll
  for (int kc = 0; kc < 4; ++kc) bf[kc] = pack_b_f32(wrow + kc * 32 + hi * 16);
  float bo = b_out[col];

  for (int mt = 0; mt * 16 < Tv; ++mt) {
    int arow = mt * 16 + n; if (arow >= Tv) arow = Tv - 1;
    v8f acc = {};
#pragma unroll
    for (int kc = 0; kc < 4; ++kc) {
      v16h a = pack_a_f16(codeh + arow * 128 + kc * 32 + hi * 16);
      acc = WMMA_F16(a, bf[kc], acc);
    }
#pragma unroll
    for (int r = 0; r < 8; ++r) {
      int mg = mt * 16 + r + 8 * hi;
      if (mg < Tv) out[(long)mg * S + col] = acc[r] + bo;
    }
  }
}

// ---------- launch ----------
extern "C" void kernel_launch(void* const* d_in, const int* in_sizes, int n_in,
                              void* d_out, int out_size, void* d_ws, size_t ws_size,
                              hipStream_t stream) {
  const int*   type_batch   = (const int*)  d_in[0];
  const int*   token_batch  = (const int*)  d_in[1];
  const int*   node_indices = (const int*)  d_in[2];
  const float* eta_t        = (const float*)d_in[3];
  const float* eta_l        = (const float*)d_in[4];
  const float* eta_r        = (const float*)d_in[5];
  const int*   tree_indices = (const int*)  d_in[6];
  const float* emb_type     = (const float*)d_in[7];
  const float* emb_token    = (const float*)d_in[8];
  const float* W_enc        = (const float*)d_in[9];
  const float* b_enc        = (const float*)d_in[10];
  const float* w_t          = (const float*)d_in[11];
  const float* w_l          = (const float*)d_in[12];
  const float* w_r          = (const float*)d_in[13];
  const float* bias_conv    = (const float*)d_in[14];
  const float* alpha        = (const float*)d_in[15];
  const float* W_out        = (const float*)d_in[16];
  const float* b_out        = (const float*)d_in[17];

  const int N   = in_sizes[0];
  const int NNv = in_sizes[6];
  const int S   = in_sizes[17];
  const int Tv  = out_size / S;

  // workspace layout
  char* ws = (char*)d_ws;
  float* pre = (float*)ws;
  size_t off = (size_t)NNv * 128 * 4;
  float* scores = (float*)(ws + off); off += (size_t)NNv * 4;
  off = (off + 255) & ~(size_t)255;
  unsigned* treeMax = (unsigned*)(ws + off); off += (size_t)Tv * 4;
  float* denom   = (float*)(ws + off);       off += (size_t)Tv * 4;
  float* codeAcc = (float*)(ws + off);       off += (size_t)Tv * 128 * 4;
  _Float16* codeh = (_Float16*)(ws + off);

  // zero accumulators (pre; then treeMax|denom|codeAcc contiguous block)
  kzero<<<2048, 256, 0, stream>>>(pre, NNv * 128);
  kzero<<<8, 256, 0, stream>>>((float*)treeMax, Tv * 130);

  // stage 1: encoder + eta-weighted conv + scatter add
  int blocks1 = (N + 127) / 128;
  size_t smem1 = 65536 + 98304 + 32768;  // 192 KB dynamic LDS
  k1_conv<<<blocks1, 256, smem1, stream>>>(type_batch, token_batch, node_indices,
                                           eta_t, eta_l, eta_r, emb_type, emb_token,
                                           W_enc, b_enc, w_t, w_l, w_r, pre, N);

  // stage 2: tanh + alpha scores + per-tree max
  k2_act<<<(NNv + 7) / 8, 256, 0, stream>>>(pre, scores, treeMax, tree_indices,
                                            alpha, bias_conv, NNv);

  // stage 3: softmax-weighted pooling accumulation
  k3_pool<<<(NNv + 7) / 8, 256, 0, stream>>>(pre, scores, treeMax, denom, codeAcc,
                                             tree_indices, NNv);

  // stage 4: normalize -> f16 code vectors
  k4_norm<<<(Tv * 128 + 255) / 256, 256, 0, stream>>>(codeAcc, denom, codeh, Tv * 128);

  // stage 5: logits GEMM
  k5_logits<<<(S / 16 + 7) / 8, 256, 0, stream>>>(codeh, W_out, b_out, (float*)d_out, S, Tv);
}
